// GCNEncoder_53515292508661
// MI455X (gfx1250) — matrix operand order repaired and checked by reading, not compiled, not run on hardware
//
#include <hip/hip_runtime.h>

#define N_NODES 50000
#define N_EDGES 320000
#define MPAD    50048           // N_NODES rounded up to 128
#define LSTR    40              // LDS row stride in u16 (80B: 16B-aligned, conflict-free)

typedef __attribute__((ext_vector_type(16))) __bf16 v16bf;
typedef __attribute__((ext_vector_type(8)))  float  v8f;

__device__ __forceinline__ unsigned short f2bf(float f) {
    unsigned int u = __builtin_bit_cast(unsigned int, f);
    u += 0x7FFFu + ((u >> 16) & 1u);          // round to nearest even
    return (unsigned short)(u >> 16);
}
__device__ __forceinline__ unsigned int pk2bf(float lo, float hi) {
    return (unsigned int)f2bf(lo) | ((unsigned int)f2bf(hi) << 16);
}

// ---------------------------------------------------------------------------
// Degree / normalization
// ---------------------------------------------------------------------------
__global__ void deg_init_k(float* deg) {
    int n = blockIdx.x * 256 + threadIdx.x;
    if (n < N_NODES) deg[n] = 1.0f;           // +1 self loop
}
__global__ void deg_count_k(const long long* __restrict__ dst, float* deg) {
    int e = blockIdx.x * 256 + threadIdx.x;
    if (e < N_EDGES) atomicAdd(&deg[dst[e]], 1.0f);
}
__global__ void deg_rsqrt_k(float* deg) {
    int n = blockIdx.x * 256 + threadIdx.x;
    if (n < N_NODES) deg[n] = rsqrtf(deg[n]);
}

// ---------------------------------------------------------------------------
// f32 [M x K] -> bf16 [MPAD x K], rows >= M zero-filled. 4 elems / thread.
// ---------------------------------------------------------------------------
__global__ void conv_a_k(const float* __restrict__ A, unsigned short* __restrict__ Ab,
                         size_t valid, size_t totalPad) {
    size_t idx = ((size_t)blockIdx.x * 256 + threadIdx.x) * 4;
    if (idx >= totalPad) return;
    float4 v = make_float4(0.f, 0.f, 0.f, 0.f);
    if (idx < valid) v = *(const float4*)(A + idx);
    uint2 p;
    p.x = pk2bf(v.x, v.y);
    p.y = pk2bf(v.z, v.w);
    *(uint2*)(Ab + idx) = p;
}

// ---------------------------------------------------------------------------
// W f32 [K x N] -> transposed bf16 Wt [N x K] (coalesced read, scattered b16)
// ---------------------------------------------------------------------------
__global__ void conv_wt_k(const float* __restrict__ W, unsigned short* __restrict__ Wt,
                          int K, int N) {
    size_t e = (size_t)blockIdx.x * 256 + threadIdx.x;
    if (e >= (size_t)K * N) return;
    int k = (int)(e / N);
    int n = (int)(e % N);
    Wt[(size_t)n * K + k] = f2bf(W[e]);
}

// ---------------------------------------------------------------------------
// bf16 WMMA GEMM: out[M,N] = Ab[MPAD,K] @ Wt[N,K]^T, f32 accumulate.
// block = 256 threads (8 wave32), tile 128x128, BK=32.
// wave w: rows (w&1)*64..+63, cols (w>>1)*32..+31  -> 8 WMMAs / K-step.
// ---------------------------------------------------------------------------
__launch_bounds__(256)
__global__ void gemm_bf16_wmma_k(const unsigned short* __restrict__ Ab,
                                 const unsigned short* __restrict__ Wt,
                                 float* __restrict__ out,
                                 int M, int K, int N) {
    __shared__ unsigned short lA[128 * LSTR];   // 128 rows x 32 bf16
    __shared__ unsigned short lB[128 * LSTR];   // 128 cols x 32 bf16 (pre-transposed)

    const int tid  = threadIdx.x;
    const int lane = tid & 31;
    const int wave = tid >> 5;
    const int wm   = wave & 1;                  // 64-row slab
    const int wn   = wave >> 1;                 // 32-col slab (0..3)
    const int rowBase = blockIdx.x * 128;
    const int colBase = blockIdx.y * 128;

    v8f acc[4][2] = {};

    // staging map: 512 segments of 8 bf16 (16B) per tile; 2 segs / thread
    const int seg0  = tid;            // + 256
    const int h4a   = (lane >> 4) * 4;          // A frag u32 offset
    const int h8b   = (lane >> 4) * 8;          // unused (B frag also uses h4a)
    const int fr    = lane & 15;

    for (int k0 = 0; k0 < K; k0 += 32) {
#pragma unroll
        for (int i = 0; i < 2; ++i) {
            int seg = seg0 + 256 * i;           // 0..511
            int r   = seg >> 2;                 // row/col within tile
            int sc  = (seg & 3) * 8;            // bf16 offset within 32-wide row
            uint4 va = *(const uint4*)(Ab + (size_t)(rowBase + r) * K + k0 + sc);
            uint4 vb = *(const uint4*)(Wt + (size_t)(colBase + r) * K + k0 + sc);
            *(uint4*)(lA + r * LSTR + sc) = va;
            *(uint4*)(lB + r * LSTR + sc) = vb;
        }
        __syncthreads();

        union Frag { unsigned int u[8]; v16bf v; } af[4], bf[2];
#pragma unroll
        for (int s = 0; s < 4; ++s) {
            const unsigned int* pa = (const unsigned int*)(lA + (wm * 64 + s * 16 + fr) * LSTR);
#pragma unroll
            for (int j = 0; j < 8; ++j)
                af[s].u[j] = pa[(j < 4 ? j : j + 4) + h4a];   // ISA 16-bit A 16x32 layout
        }
#pragma unroll
        for (int t = 0; t < 2; ++t) {
            const unsigned int* pb = (const unsigned int*)(lB + (wn * 32 + t * 16 + fr) * LSTR);
#pragma unroll
            for (int j = 0; j < 8; ++j)
                bf[t].u[j] = pb[(j < 4 ? j : j + 4) + h4a];   // ISA 16-bit B 32x16 layout
        }
        __syncthreads();

#pragma unroll
        for (int s = 0; s < 4; ++s)
#pragma unroll
            for (int t = 0; t < 2; ++t)
                acc[s][t] = __builtin_amdgcn_wmma_f32_16x16x32_bf16(
                    false, af[s].v, false, bf[t].v, (short)0, acc[s][t], false, false);
    }

    // C/D layout: VGPR r -> lanes 0-15: (M=r, N=lane); lanes 16-31: (M=8+r, N=lane-16)
    const int srow = (lane >> 4) * 8;
#pragma unroll
    for (int s = 0; s < 4; ++s) {
#pragma unroll
        for (int t = 0; t < 2; ++t) {
            int gc = colBase + wn * 32 + t * 16 + fr;
#pragma unroll
            for (int r = 0; r < 8; ++r) {
                int gr = rowBase + wm * 64 + s * 16 + srow + r;
                if (gr < M) out[(size_t)gr * N + gc] = acc[s][t][r];
            }
        }
    }
}

// ---------------------------------------------------------------------------
// agg[n,c] = hW[n,c] * dinv[n]^2 + b[c]   (self-loop + bias init)
// ---------------------------------------------------------------------------
__global__ void agg_init_k(const float* __restrict__ hW,
                           const float* __restrict__ dinv,
                           const float* __restrict__ bias,
                           float* __restrict__ agg, int C) {
    size_t idx = (size_t)blockIdx.x * 256 + threadIdx.x;
    if (idx >= (size_t)N_NODES * C) return;
    int n = (int)(idx / C);
    int c = (int)(idx % C);
    float di = dinv[n];
    agg[idx] = hW[idx] * di * di + bias[c];
}

// ---------------------------------------------------------------------------
// per-edge scatter: agg[dst] += hW[src] * dinv[src]*dinv[dst]
// ---------------------------------------------------------------------------
__launch_bounds__(128)
__global__ void edge_scatter_k(const long long* __restrict__ ei,
                               const float* __restrict__ hW,
                               const float* __restrict__ dinv,
                               float* __restrict__ agg, int C) {
    int e = blockIdx.x;
    int c = blockIdx.y * 128 + threadIdx.x;
    long long s = ei[e];
    long long d = ei[N_EDGES + e];
    float norm = dinv[s] * dinv[d];
    float v = hW[(size_t)s * C + c] * norm;
    atomicAdd(&agg[(size_t)d * C + c], v);
}

__global__ void relu_k(float* h, size_t total) {
    size_t idx = (size_t)blockIdx.x * 256 + threadIdx.x;
    if (idx < total) h[idx] = fmaxf(h[idx], 0.0f);
}

// ---------------------------------------------------------------------------
extern "C" void kernel_launch(void* const* d_in, const int* in_sizes, int n_in,
                              void* d_out, int out_size, void* d_ws, size_t ws_size,
                              hipStream_t stream) {
    (void)in_sizes; (void)n_in; (void)out_size; (void)ws_size;

    const float*     x  = (const float*)d_in[0];
    const long long* ei = (const long long*)d_in[1];   // [2, N_EDGES] int64
    const float* Wts[5] = {(const float*)d_in[2], (const float*)d_in[4],
                           (const float*)d_in[6], (const float*)d_in[8],
                           (const float*)d_in[10]};
    const float* bs[5]  = {(const float*)d_in[3], (const float*)d_in[5],
                           (const float*)d_in[7], (const float*)d_in[9],
                           (const float*)d_in[11]};
    const int dims[6] = {512, 2048, 1024, 512, 256, 128};

    char* ws = (char*)d_ws;
    size_t off = 0;
    float* dinv = (float*)(ws + off);
    off += ((size_t)N_NODES * 4 + 255) & ~(size_t)255;
    float* B0 = (float*)(ws + off);              // agg / next h   (f32, N x 2048 max)
    off += (size_t)N_NODES * 2048 * 4;
    float* B1 = (float*)(ws + off);              // hW             (f32, N x 2048 max)
    off += (size_t)N_NODES * 2048 * 4;
    unsigned short* Abf = (unsigned short*)(ws + off);   // bf16 [MPAD x 2048 max]
    off += (size_t)MPAD * 2048 * 2;
    unsigned short* Wtb = (unsigned short*)(ws + off);   // bf16 transposed W (<= 4MB)

    // degree -> dinv
    int nb = (N_NODES + 255) / 256;
    deg_init_k<<<nb, 256, 0, stream>>>(dinv);
    deg_count_k<<<(N_EDGES + 255) / 256, 256, 0, stream>>>(ei + N_EDGES, dinv);
    deg_rsqrt_k<<<nb, 256, 0, stream>>>(dinv);

    const float* h = x;
    for (int L = 0; L < 5; ++L) {
        int K = dims[L];
        int C = dims[L + 1];
        float* hW  = B1;
        float* agg = (L == 4) ? (float*)d_out : B0;

        size_t valid    = (size_t)N_NODES * K;
        size_t totalPad = (size_t)MPAD * K;
        conv_a_k<<<(unsigned)((totalPad / 4 + 255) / 256), 256, 0, stream>>>(h, Abf, valid, totalPad);
        conv_wt_k<<<(unsigned)(((size_t)K * C + 255) / 256), 256, 0, stream>>>(Wts[L], Wtb, K, C);

        dim3 gg(MPAD / 128, C / 128);
        gemm_bf16_wmma_k<<<gg, 256, 0, stream>>>(Abf, Wtb, hW, N_NODES, K, C);

        size_t total = (size_t)N_NODES * C;
        agg_init_k<<<(unsigned)((total + 255) / 256), 256, 0, stream>>>(hW, dinv, bs[L], agg, C);

        dim3 ge(N_EDGES, C / 128);
        edge_scatter_k<<<ge, 128, 0, stream>>>(ei, hW, dinv, agg, C);

        if (L != 4)
            relu_k<<<(unsigned)((total + 255) / 256), 256, 0, stream>>>(agg, total);

        h = agg;
    }
}
